// SelfAtt_36636071035664
// MI455X (gfx1250) — matrix operand order repaired and checked by reading, not compiled, not run on hardware
//
#include <hip/hip_runtime.h>

// ---------------------------------------------------------------------------
// Self-attention for (1, 32, 256, 16, 16) fp32 input, H = dk = dv = 256,
// N = 8192 tokens.  Heavy math in v_wmma_f32_16x16x32_bf16 (fp32 accum),
// tile staging via global_load_async_to_lds_b128 (ASYNCcnt).
// ---------------------------------------------------------------------------

typedef __attribute__((ext_vector_type(16))) __bf16 v16bf;
typedef __attribute__((ext_vector_type(8)))  float  v8f;

union Frag16 { v16bf v; unsigned int u[8]; };

static constexpr int N_TOK = 8192;   // 32 * 16 * 16
static constexpr int HD    = 256;    // hidden = dk = dv

__device__ __forceinline__ unsigned short f2bf(float f) {
  unsigned int x = __float_as_uint(f);
  x += 0x7fffu + ((x >> 16) & 1u);           // round-to-nearest-even
  return (unsigned short)(x >> 16);
}

// K pair offsets for the 16x32 bf16 A-matrix VGPR layout (ISA 7.12.2)
__device__ __forceinline__ int a_kb(int v, int hf) {
  return ((v < 4) ? 2 * v : 16 + 2 * (v - 4)) + 8 * hf;
}
// K pair offsets for the 32x16 bf16 B-matrix VGPR layout
__device__ __forceinline__ int b_kb(int v, int hf) {
  return ((v < 4) ? 2 * v : 16 + 2 * (v - 4)) + 8 * hf;
}

// Async 16-byte copy global -> LDS (CDNA5, tracked by ASYNCcnt).
__device__ __forceinline__ void async_g2l_b128(const void* lds_ptr, const void* gptr) {
  unsigned int lds_off = (unsigned int)(unsigned long long)lds_ptr; // addr[31:0] = LDS offset
  asm volatile("global_load_async_to_lds_b128 %0, %1, off"
               :: "v"(lds_off), "v"((unsigned long long)gptr)
               : "memory");
}
__device__ __forceinline__ void wait_async0() {
  asm volatile("s_wait_asynccnt 0x0" ::: "memory");
}

// ---------------------------------------------------------------------------
// Kernel 0: transpose + convert weights: W (fp32, [c][d]) -> Wt (bf16, [d][c])
// ---------------------------------------------------------------------------
__global__ __launch_bounds__(256)
void wtrans_kernel(const float* __restrict__ Wq, const float* __restrict__ Wk,
                   const float* __restrict__ Wv, unsigned short* __restrict__ Wt) {
  const int c = blockIdx.x;
  const int m = blockIdx.y;
  const int d = threadIdx.x;
  const float* W = (m == 0) ? Wq : ((m == 1) ? Wk : Wv);
  Wt[m * 65536 + d * 256 + c] = f2bf(W[c * 256 + d]);
}

// ---------------------------------------------------------------------------
// Kernel A: QKV projection.
//   Qb, Kb : bf16 row-major [n][d] ;  Vt : bf16 row-major [d][n]
// One block = one 16-token tile; 8 waves x 6 output 16-col tiles (3*256 cols).
// ---------------------------------------------------------------------------
__global__ __launch_bounds__(256)
void qkv_proj_kernel(const float* __restrict__ hs,
                     const unsigned short* __restrict__ Wt,
                     unsigned short* __restrict__ Qb,
                     unsigned short* __restrict__ Kb,
                     unsigned short* __restrict__ Vt) {
  __shared__ unsigned short sX[16 * 272];
  const int tid = threadIdx.x;
  const int n0  = blockIdx.x * 16;
  const int t   = n0 >> 8;
  const int rc0 = n0 & 255;
  const float* xb = hs + (size_t)t * 65536 + rc0;

  // Stage X tile -> bf16 LDS (transposing feature-major -> token-major)
  {
    const int w  = tid & 15;
    const int cb = tid >> 4;
#pragma unroll
    for (int p = 0; p < 16; ++p) {
      const int c = p * 16 + cb;
      sX[w * 272 + c] = f2bf(xb[c * 256 + w]);
    }
  }
  __syncthreads();

  const int lane = tid & 31, wid = tid >> 5;
  const int hf = lane >> 4, mr = lane & 15, nc = lane & 15;

  Frag16 qf[8];
#pragma unroll
  for (int kc = 0; kc < 8; ++kc)
#pragma unroll
    for (int v = 0; v < 8; ++v)
      qf[kc].u[v] = *(const unsigned int*)&sX[mr * 272 + kc * 32 + a_kb(v, hf)];

#pragma unroll
  for (int tt = 0; tt < 6; ++tt) {
    const int gcol = (wid * 6 + tt) * 16;            // 0..767, wave-uniform
    const int mat  = gcol >> 8;                      // 0=Q 1=K 2=V
    const int dcol = gcol & 255;
    const unsigned short* W = Wt + mat * 65536;
    v8f acc;
#pragma unroll
    for (int v = 0; v < 8; ++v) acc[v] = 0.f;
#pragma unroll
    for (int kc = 0; kc < 8; ++kc) {
      Frag16 bf;
#pragma unroll
      for (int v = 0; v < 8; ++v)
        bf.u[v] = *(const unsigned int*)&W[(dcol + nc) * 256 + kc * 32 + b_kb(v, hf)];
      acc = __builtin_amdgcn_wmma_f32_16x16x32_bf16(
          false, qf[kc].v, false, bf.v, (short)0, acc, false, false);
    }
    // Branch-free store: uniform pointer + stride selects (no exec divergence)
    unsigned short* dst = (mat == 0) ? Qb : ((mat == 1) ? Kb : Vt);
    const size_t row_s = (mat == 2) ? (size_t)1 : (size_t)256;      // stride of token index
    const size_t col_s = (mat == 2) ? (size_t)N_TOK : (size_t)1;    // stride of d index
#pragma unroll
    for (int v = 0; v < 8; ++v) {
      const int m = v + 8 * hf;
      dst[(size_t)(n0 + m) * row_s + (size_t)(dcol + nc) * col_s] = f2bf(acc[v]);
    }
  }
}

// ---------------------------------------------------------------------------
// Kernel B: flash attention.  Block = 32 queries (2 q-tiles), 8 waves.
// wave -> (q-tile = wid>>2, 64 dv cols = (wid&3)*64).  Key chunks of 32.
// K/V/Q tiles staged with global_load_async_to_lds_b128.
// ---------------------------------------------------------------------------
__global__ __launch_bounds__(256)
void attn_kernel(const unsigned short* __restrict__ Qb,
                 const unsigned short* __restrict__ Kb,
                 const unsigned short* __restrict__ Vt,
                 float* __restrict__ out) {
  __shared__ __align__(16) unsigned short smem[27648];   // 55296 B
  unsigned short* sQ = smem;                 // 32 x 272  (reused as P buffers)
  unsigned short* sK = smem + 8704;          // 32 x 272  (row = 544 B, 16B-aligned)
  unsigned short* sV = smem + 17408;         // 256 x 40  (row =  80 B, 16B-aligned)

  const int tid  = threadIdx.x;
  const int lane = tid & 31, wid = tid >> 5;
  const int hf = lane >> 4, mr = lane & 15, ncl = lane & 15;
  const int qt = wid >> 2, dg = wid & 3;
  const int n0b = blockIdx.x * 32;
  const int n0  = n0b + qt * 16;
  const int dbase = dg * 64;

  // Stage Q (32 tokens x 256) via async copy: 1024 x 16B
#pragma unroll
  for (int i = 0; i < 4; ++i) {
    const int j = i * 256 + tid;
    const int tok = j >> 5, cc = j & 31;                 // 32 x 16B per row
    async_g2l_b128(&sQ[tok * 272 + cc * 8],
                   Qb + (size_t)(n0b + tok) * 256 + cc * 8);
  }
  wait_async0();
  __syncthreads();

  Frag16 qf[8];
#pragma unroll
  for (int kc = 0; kc < 8; ++kc)
#pragma unroll
    for (int v = 0; v < 8; ++v)
      qf[kc].u[v] =
          *(const unsigned int*)&sQ[(qt * 16 + mr) * 272 + kc * 32 + a_kb(v, hf)];

  v8f acc[4];
#pragma unroll
  for (int dt = 0; dt < 4; ++dt)
#pragma unroll
    for (int v = 0; v < 8; ++v) acc[dt][v] = 0.f;
  float mrow[8], lrow[8];
#pragma unroll
  for (int v = 0; v < 8; ++v) { mrow[v] = -3.0e38f; lrow[v] = 0.f; }

  unsigned short* pb = sQ + wid * 544;       // per-wave 16x32 (+pad) P buffer

  for (int kb0 = 0; kb0 < N_TOK; kb0 += 32) {
    __syncthreads();
    // Stage K chunk (32 x 256) and Vt chunk (256 x 32) : 2048 x 16B async
#pragma unroll
    for (int i = 0; i < 4; ++i) {
      const int j = i * 256 + tid;
      { const int tok = j >> 5, cc = j & 31;
        async_g2l_b128(&sK[tok * 272 + cc * 8],
                       Kb + (size_t)(kb0 + tok) * 256 + cc * 8); }
      { const int d = j >> 2, kk = j & 3;                // 4 x 16B per Vt row
        async_g2l_b128(&sV[d * 40 + kk * 8],
                       Vt + (size_t)d * 8192 + kb0 + kk * 8); }
    }
    // Prefetch next chunk while this one is in flight
    if (kb0 + 32 < N_TOK) {
      __builtin_prefetch(Kb + (size_t)(kb0 + 32) * 256 + tid * 32, 0, 1);
      __builtin_prefetch(Vt + (size_t)tid * 8192 + kb0 + 32, 0, 1);
    }
    wait_async0();
    __syncthreads();

    // S = Q K^T over dk=256 : two 16x16 column tiles, K-depth 32 each
    v8f s0, s1;
#pragma unroll
    for (int v = 0; v < 8; ++v) { s0[v] = 0.f; s1[v] = 0.f; }
#pragma unroll
    for (int kc = 0; kc < 8; ++kc) {
      Frag16 b0, b1;
#pragma unroll
      for (int v = 0; v < 8; ++v) {
        const int kb = kc * 32 + b_kb(v, hf);
        b0.u[v] = *(const unsigned int*)&sK[ncl * 272 + kb];
        b1.u[v] = *(const unsigned int*)&sK[(16 + ncl) * 272 + kb];
      }
      s0 = __builtin_amdgcn_wmma_f32_16x16x32_bf16(
          false, qf[kc].v, false, b0.v, (short)0, s0, false, false);
      s1 = __builtin_amdgcn_wmma_f32_16x16x32_bf16(
          false, qf[kc].v, false, b1.v, (short)0, s1, false, false);
    }

    // Online softmax (row stats per C-fragment row: M = v + 8*hf)
    float alpha[8];
#pragma unroll
    for (int v = 0; v < 8; ++v) {
      float a = fmaxf(s0[v], s1[v]);
#pragma unroll
      for (int off = 1; off < 16; off <<= 1) a = fmaxf(a, __shfl_xor(a, off, 16));
      const float mnew = fmaxf(mrow[v], a * 0.0625f);      // 1/sqrt(256)
      alpha[v] = __expf(mrow[v] - mnew);
      const float p0 = __expf(s0[v] * 0.0625f - mnew);
      const float p1 = __expf(s1[v] * 0.0625f - mnew);
      s0[v] = p0; s1[v] = p1;
      float rs = p0 + p1;
#pragma unroll
      for (int off = 1; off < 16; off <<= 1) rs += __shfl_xor(rs, off, 16);
      lrow[v] = lrow[v] * alpha[v] + rs;
      mrow[v] = mnew;
    }

    // P (C-layout) -> per-wave LDS -> A-layout fragment
#pragma unroll
    for (int v = 0; v < 8; ++v) {
      const int m = v + 8 * hf;
      pb[m * 34 + ncl]      = f2bf(s0[v]);
      pb[m * 34 + 16 + ncl] = f2bf(s1[v]);
    }
    Frag16 pf;
#pragma unroll
    for (int v = 0; v < 8; ++v)
      pf.u[v] = *(const unsigned int*)&pb[mr * 34 + a_kb(v, hf)];

    // O = alpha*O + P @ V  (4 dv tiles of 16)
#pragma unroll
    for (int dt = 0; dt < 4; ++dt) {
      Frag16 bv;
      const int d = dbase + dt * 16 + ncl;
#pragma unroll
      for (int v = 0; v < 8; ++v)
        bv.u[v] = *(const unsigned int*)&sV[d * 40 + b_kb(v, hf)];
      v8f c = acc[dt];
#pragma unroll
      for (int v = 0; v < 8; ++v) c[v] *= alpha[v];
      acc[dt] = __builtin_amdgcn_wmma_f32_16x16x32_bf16(
          false, pf.v, false, bv.v, (short)0, c, false, false);
    }
  }

  // Epilogue: normalize, scatter to (1, 32, 256, 16, 16) layout
#pragma unroll
  for (int v = 0; v < 8; ++v) {
    const int m = v + 8 * hf;
    const int n = n0 + m;
    const int t = n >> 8, rc = n & 255;
    const float inv = 1.f / lrow[v];
#pragma unroll
    for (int dt = 0; dt < 4; ++dt) {
      const int d = dbase + dt * 16 + ncl;
      out[(size_t)t * 65536 + (size_t)d * 256 + rc] = acc[dt][v] * inv;
    }
  }
}

// ---------------------------------------------------------------------------
// Launch
// ---------------------------------------------------------------------------
extern "C" void kernel_launch(void* const* d_in, const int* in_sizes, int n_in,
                              void* d_out, int out_size, void* d_ws, size_t ws_size,
                              hipStream_t stream) {
  (void)in_sizes; (void)n_in; (void)out_size; (void)ws_size;
  const float* hs = (const float*)d_in[0];
  const float* Wq = (const float*)d_in[1];
  const float* Wk = (const float*)d_in[2];
  const float* Wv = (const float*)d_in[3];

  char* ws = (char*)d_ws;
  unsigned short* Wt = (unsigned short*)ws;                       // 3*256*256 bf16
  unsigned short* Qb = (unsigned short*)(ws + 393216);            // N x 256 bf16
  unsigned short* Kb = Qb + (size_t)N_TOK * HD;                   // N x 256 bf16
  unsigned short* Vt = Kb + (size_t)N_TOK * HD;                   // 256 x N bf16
  float* out = (float*)d_out;

  wtrans_kernel<<<dim3(256, 3), 256, 0, stream>>>(Wq, Wk, Wv, Wt);
  qkv_proj_kernel<<<N_TOK / 16, 256, 0, stream>>>(hs, Wt, Qb, Kb, Vt);
  attn_kernel<<<N_TOK / 32, 256, 0, stream>>>(Qb, Kb, Vt, out);
}
